// _F3Block_42588895707390
// MI455X (gfx1250) — matrix operand order repaired and checked by reading, not compiled, not run on hardware
//
#include <hip/hip_runtime.h>
#include <math.h>

// ---------------------------------------------------------------------------
// Types for CDNA5 WMMA (wave32, 16x16x32 bf16 -> f32 accumulate)
// ---------------------------------------------------------------------------
typedef __attribute__((ext_vector_type(16))) __bf16       bf16x16;
typedef __attribute__((ext_vector_type(8)))  float        f32x8;
typedef __attribute__((ext_vector_type(4)))  unsigned int u32x4;

union FragU { u32x4 u[2]; bf16x16 v; };

#define EMBED_DIM 768
#define N_HEADS   12
#define HEAD_DIM  64
#define T_HEAD    32
#define T_DIM     384
#define SEQ       2048
#define BATCH     2
#define NROWS     (BATCH * SEQ)     // 4096
#define HID4      (EMBED_DIM * 4)   // 3072

__device__ __forceinline__ unsigned short f2bf(float f) {
  union { float f; unsigned u; } c; c.f = f;
  unsigned u = c.u;
  return (unsigned short)((u + 0x7FFFu + ((u >> 16) & 1u)) >> 16); // RNE
}

__device__ __forceinline__ f32x8 wmma_bf16(const FragU& a, const FragU& b, f32x8 c) {
  return __builtin_amdgcn_wmma_f32_16x16x32_bf16(false, a.v, false, b.v,
                                                 (short)0, c, false, false);
}

__device__ __forceinline__ float gelu_tanh(float x) {
  return 0.5f * x * (1.0f + tanhf(0.7978845608028654f * (x + 0.044715f * x * x * x)));
}

// ---------------------------------------------------------------------------
// CDNA5 async global->LDS DMA (ASYNCcnt path; ISA 15.18.3 / 08_async_tensor 4.4)
//   IOFFSET is added to BOTH the LDS (VDST) and global (VADDR) addresses, so a
//   burst shares one base pair and uses offset: immediates -> no v_add_u64s.
// ---------------------------------------------------------------------------
__device__ __forceinline__ void async_b128_x4(unsigned lds_addr, const void* gaddr) {
  asm volatile("global_load_async_to_lds_b128 %0, %1, off\n\t"
               "global_load_async_to_lds_b128 %0, %1, off offset:16\n\t"
               "global_load_async_to_lds_b128 %0, %1, off offset:32\n\t"
               "global_load_async_to_lds_b128 %0, %1, off offset:48"
               :: "v"(lds_addr), "v"(gaddr) : "memory");
}
__device__ __forceinline__ void async_b128_x2(unsigned lds_addr, const void* gaddr) {
  asm volatile("global_load_async_to_lds_b128 %0, %1, off\n\t"
               "global_load_async_to_lds_b128 %0, %1, off offset:16"
               :: "v"(lds_addr), "v"(gaddr) : "memory");
}
__device__ __forceinline__ void wait_async0() {
  asm volatile("s_wait_asynccnt 0" ::: "memory");
}
__device__ __forceinline__ unsigned lds_lo32(const void* p) {
  return (unsigned)(size_t)p;  // flat addr[31:0] == LDS byte address (aperture rule)
}

// ---------------------------------------------------------------------------
// fp32 -> bf16 converters (run once per launch; traffic negligible, L2-resident)
// ---------------------------------------------------------------------------
__global__ void k_f32_to_bf16(const float* __restrict__ in,
                              unsigned short* __restrict__ out, int n) {
  for (int i = blockIdx.x * blockDim.x + threadIdx.x; i < n; i += gridDim.x * blockDim.x)
    out[i] = f2bf(in[i]);
}

// Wt[n*K + k] = bf16(W[k*N + n])  (weights stored transposed -> contiguous B frags)
__global__ void k_transpose_bf16(const float* __restrict__ in,
                                 unsigned short* __restrict__ out, int K, int N) {
  int total = K * N;
  for (int i = blockIdx.x * blockDim.x + threadIdx.x; i < total; i += gridDim.x * blockDim.x) {
    int k = i / N, n = i - k * N;
    out[(size_t)n * K + k] = f2bf(in[i]);
  }
}

// ---------------------------------------------------------------------------
// Tiled bf16 WMMA GEMM: C[M][N] = A[M][K] @ Wt[N][K]^T + bias, fused epilogues
//   block = 256 threads (8 waves), tile 128x64, BK=64
//   Double-buffered LDS filled by async DMA; 8 WMMA per wave per buffer.
// ---------------------------------------------------------------------------
enum { MODE_Q = 0, MODE_KV = 1, MODE_RES = 2, MODE_GELU = 3 };

template <int MODE>
__global__ __launch_bounds__(256) void k_gemm_bf16(
    const unsigned short* __restrict__ A,    // [M][K] bf16
    const unsigned short* __restrict__ Wt,   // [N][K] bf16 (transposed weight)
    const float* __restrict__ bias,          // [N]
    const float* __restrict__ res,           // [M][N] f32 residual (MODE_RES)
    float* __restrict__ outf,                // f32 out (MODE_RES)
    unsigned short* __restrict__ outb,       // bf16 out (other modes)
    int M, int N, int K) {
  constexpr int BM = 128, BN = 64, BK = 64, LDT = 72;  // pad keeps 16B align
  __shared__ alignas(16) unsigned short As[2][BM * LDT];
  __shared__ alignas(16) unsigned short Bs[2][BN * LDT];
  (void)M;

  const int tid  = threadIdx.x;
  const int lane = tid & 31;
  const int w    = tid >> 5;   // 0..7
  const int wm   = w >> 1;     // 0..3
  const int wn   = w & 1;      // 0..1
  const int bm   = blockIdx.y * BM;
  const int bn   = blockIdx.x * BN;

  f32x8 acc[2][2];
  for (int i = 0; i < 2; i++)
    for (int j = 0; j < 2; j++) acc[i][j] = f32x8{};

  // Fragment gather geometry (ISA 7.12.2 layouts)
  const int frow = lane & 15;
  const int akb  = (lane < 16) ? 0 : 8;    // A: K chunks {0..7,16..23}/{8..15,24..31}
  const int bkb  = (lane < 16) ? 0 : 8;    // B: K chunks {0..7,16..23}/{8..15,24..31}

  // Staging geometry: A = 32 u16/thread (4 x b128), B = 16 u16/thread (2 x b128)
  const int sa_row = tid >> 1;             // 0..127
  const int sa_k   = (tid & 1) * 32;
  const int sb_row = tid >> 2;             // 0..63
  const int sb_k   = (tid & 3) * 16;
  const unsigned short* Aptr = A  + (size_t)(bm + sa_row) * K + sa_k;
  const unsigned short* Bptr = Wt + (size_t)(bn + sb_row) * K + sb_k;

  auto stage = [&](int buf, int k0) {
    async_b128_x4(lds_lo32(&As[buf][sa_row * LDT + sa_k]), Aptr + k0);
    async_b128_x2(lds_lo32(&Bs[buf][sb_row * LDT + sb_k]), Bptr + k0);
  };

  auto compute = [&](int buf) {
    for (int kk = 0; kk < BK; kk += 32) {
      FragU af[2], bf[2];
      for (int sm = 0; sm < 2; sm++) {
        int base = (wm * 32 + sm * 16 + frow) * LDT + kk;
        af[sm].u[0] = *(const u32x4*)&As[buf][base + akb];
        af[sm].u[1] = *(const u32x4*)&As[buf][base + akb + 16];
      }
      for (int sn = 0; sn < 2; sn++) {
        int base = (wn * 32 + sn * 16 + frow) * LDT + kk + bkb;
        bf[sn].u[0] = *(const u32x4*)&Bs[buf][base];
        bf[sn].u[1] = *(const u32x4*)&Bs[buf][base + 16];
      }
      acc[0][0] = wmma_bf16(af[0], bf[0], acc[0][0]);
      acc[0][1] = wmma_bf16(af[0], bf[1], acc[0][1]);
      acc[1][0] = wmma_bf16(af[1], bf[0], acc[1][0]);
      acc[1][1] = wmma_bf16(af[1], bf[1], acc[1][1]);
    }
  };

  // Pipeline: DMA of tile k+1 overlaps WMMA on tile k; one barrier per BK=64.
  stage(0, 0);
  wait_async0();
  __syncthreads();
  int buf = 0;
  for (int k0 = 0; k0 < K; k0 += BK) {
    const int next = k0 + BK;
    if (next < K) stage(buf ^ 1, next);
    compute(buf);
    if (next < K) wait_async0();
    __syncthreads();
    buf ^= 1;
  }

  // Epilogue: C/D layout -> row = r + (lane>=16 ? 8 : 0), col = lane&15
  const int rr = (lane >= 16) ? 8 : 0;
  for (int sm = 0; sm < 2; sm++)
    for (int sn = 0; sn < 2; sn++) {
      int col = bn + wn * 32 + sn * 16 + (lane & 15);
      float bcol = bias[col];
      for (int r = 0; r < 8; r++) {
        int row = bm + wm * 32 + sm * 16 + r + rr;
        float v = acc[sm][sn][r] + bcol;
        if (MODE == MODE_Q) {
          // fold Lorentz eta (+/-1 on time dims) and 1/sqrt(HEAD_DIM) into q
          int dh = col & (HEAD_DIM - 1);
          v *= (dh < T_HEAD) ? -0.125f : 0.125f;
          outb[(size_t)row * N + col] = f2bf(v);
        } else if (MODE == MODE_KV) {
          outb[(size_t)row * N + col] = f2bf(v);
        } else if (MODE == MODE_GELU) {
          outb[(size_t)row * N + col] = f2bf(gelu_tanh(v));
        } else {  // MODE_RES
          v += res[(size_t)row * N + col];
          outf[(size_t)row * N + col] = v;
        }
      }
    }
}

// ---------------------------------------------------------------------------
// Flash attention: block = 128 threads (4 waves) = 64 queries, one (b,h)
//   K tile staged by async DMA; V staged through VGPRs (needs transpose).
//   eta and 1/8 scale already folded into q_bf
// ---------------------------------------------------------------------------
__global__ __launch_bounds__(128) void k_flash(
    const unsigned short* __restrict__ Q,
    const unsigned short* __restrict__ Kb,
    const unsigned short* __restrict__ Vb,
    unsigned short* __restrict__ O) {
  constexpr int LDT = 72;
  __shared__ alignas(16) unsigned short Ks[64 * LDT];   // [key][d] (B for logits)
  __shared__ alignas(16) unsigned short Vs[64 * LDT];   // [d][key] (B for PV)
  __shared__ float Srow[4][16][64];
  __shared__ alignas(16) unsigned short Pw[4][16 * LDT];
  __shared__ float alphas[4][16];
  __shared__ float rowm[4][16];
  __shared__ float rowl[4][16];

  const int tid  = threadIdx.x;
  const int lane = tid & 31;
  const int w    = tid >> 5;
  const int qt   = blockIdx.x;
  const int bh   = blockIdx.y;
  const int b    = bh / N_HEADS, h = bh % N_HEADS;
  const int qbase = qt * 64 + w * 16;
  const size_t headoff = (size_t)h * HEAD_DIM;

  // Q fragments for both 32-deep k-steps live in registers for the whole pass
  FragU qf[2];
  {
    int row = qbase + (lane & 15);
    const unsigned short* qp = Q + ((size_t)(b * SEQ + row)) * EMBED_DIM + headoff;
    int kb = (lane < 16) ? 0 : 8;
    for (int ks = 0; ks < 2; ks++) {
      qf[ks].u[0] = *(const u32x4*)(qp + ks * 32 + kb);
      qf[ks].u[1] = *(const u32x4*)(qp + ks * 32 + kb + 16);
    }
  }
  if (lane < 16) { rowm[w][lane] = -1e30f; rowl[w][lane] = 0.0f; }

  f32x8 of[4];
  for (int d = 0; d < 4; d++) of[d] = f32x8{};

  const int st_row = tid >> 1;        // 0..63
  const int st_c   = (tid & 1) * 32;
  const unsigned klds = lds_lo32(&Ks[st_row * LDT + st_c]);

  for (int kt = 0; kt < SEQ / 64; kt++) {
    const int kbase = kt * 64;
    {  // stage K via async DMA burst; V via VGPRs with transpose scatter
      const unsigned short* kp =
          Kb + ((size_t)(b * SEQ + kbase + st_row)) * EMBED_DIM + headoff + st_c;
      async_b128_x4(klds, kp);

      const unsigned short* vp =
          Vb + ((size_t)(b * SEQ + kbase + st_row)) * EMBED_DIM + headoff + st_c;
      union { u32x4 q[4]; unsigned short hh[32]; } tv;
      tv.q[0] = *(const u32x4*)(vp);
      tv.q[1] = *(const u32x4*)(vp + 8);
      tv.q[2] = *(const u32x4*)(vp + 16);
      tv.q[3] = *(const u32x4*)(vp + 24);
      for (int j = 0; j < 32; j++) Vs[(st_c + j) * LDT + st_row] = tv.hh[j];
      wait_async0();
    }
    __syncthreads();

    {  // logits: S(16x64) = Q(16x64) @ K^T
      const int n15 = lane & 15;
      const int rr  = (lane >= 16) ? 8 : 0;
      for (int ns = 0; ns < 4; ns++) {
        f32x8 c = f32x8{};
        for (int ks = 0; ks < 2; ks++) {
          FragU bb;
          int kb2 = ks * 32 + ((lane < 16) ? 0 : 8);
          bb.u[0] = *(const u32x4*)&Ks[(ns * 16 + n15) * LDT + kb2];
          bb.u[1] = *(const u32x4*)&Ks[(ns * 16 + n15) * LDT + kb2 + 16];
          c = wmma_bf16(qf[ks], bb, c);
        }
        for (int r = 0; r < 8; r++) Srow[w][r + rr][ns * 16 + n15] = c[r];
      }
    }
    __syncthreads();

    if (lane < 16) {  // online softmax, one query row per lane (exp on TRANS pipe)
      float m_old = rowm[w][lane];
      float mx = m_old;
      for (int j = 0; j < 64; j++) mx = fmaxf(mx, Srow[w][lane][j]);
      float lsum = 0.0f;
      for (int j = 0; j < 64; j++) {
        float p = __expf(Srow[w][lane][j] - mx);
        lsum += p;
        Pw[w][lane * LDT + j] = f2bf(p);
      }
      float al = __expf(m_old - mx);
      rowl[w][lane]   = rowl[w][lane] * al + lsum;
      rowm[w][lane]   = mx;
      alphas[w][lane] = al;
    }
    __syncthreads();

    {  // rescale accumulators, then O += P @ V
      const int rr = (lane >= 16) ? 8 : 0;
      float al[8];
      for (int r = 0; r < 8; r++) al[r] = alphas[w][r + rr];
      for (int d = 0; d < 4; d++)
        for (int r = 0; r < 8; r++) of[d][r] *= al[r];

      const int n15  = lane & 15;
      const int pakb = (lane < 16) ? 0 : 8;
      const int vbkb = (lane < 16) ? 0 : 8;
      for (int ks = 0; ks < 2; ks++) {
        FragU pa;
        pa.u[0] = *(const u32x4*)&Pw[w][n15 * LDT + ks * 32 + pakb];
        pa.u[1] = *(const u32x4*)&Pw[w][n15 * LDT + ks * 32 + pakb + 16];
        for (int d = 0; d < 4; d++) {
          FragU vv;
          vv.u[0] = *(const u32x4*)&Vs[(d * 16 + n15) * LDT + ks * 32 + vbkb];
          vv.u[1] = *(const u32x4*)&Vs[(d * 16 + n15) * LDT + ks * 32 + vbkb + 16];
          of[d] = wmma_bf16(pa, vv, of[d]);
        }
      }
    }
    __syncthreads();
  }

  {  // finalize: O /= rowsum, emit bf16 for the O-projection GEMM
    const int rr = (lane >= 16) ? 8 : 0;
    float li[8];
    for (int r = 0; r < 8; r++) li[r] = 1.0f / rowl[w][r + rr];
    const int n15 = lane & 15;
    for (int d = 0; d < 4; d++)
      for (int r = 0; r < 8; r++) {
        int row = qbase + r + rr;
        O[((size_t)(b * SEQ + row)) * EMBED_DIM + headoff + d * 16 + n15] =
            f2bf(of[d][r] * li[r]);
      }
  }
}

// ---------------------------------------------------------------------------
// Minkowski LN: two independent 384-wide LayerNorms per row; 128 threads/row
// ---------------------------------------------------------------------------
__device__ __forceinline__ float blk_reduce(float v, float* red, int tid) {
  red[tid] = v;
  __syncthreads();
  for (int s = 64; s > 0; s >>= 1) {
    if (tid < s) red[tid] += red[tid + s];
    __syncthreads();
  }
  float r = red[0];
  __syncthreads();
  return r;
}

__global__ __launch_bounds__(128) void k_minkln(
    const float* __restrict__ in,
    const float* __restrict__ gt, const float* __restrict__ bt,
    const float* __restrict__ gs, const float* __restrict__ bs,
    float* __restrict__ outf, unsigned short* __restrict__ outb) {
  __shared__ float red[128];
  const int tid = threadIdx.x;
  const size_t row = blockIdx.x;
  const float* xr = in + row * EMBED_DIM;

  float vt[3], vs[3];
  for (int i = 0; i < 3; i++) {
    vt[i] = xr[tid + i * 128];
    vs[i] = xr[T_DIM + tid + i * 128];
  }
  float sum_t = blk_reduce(vt[0] + vt[1] + vt[2], red, tid);
  float sum_s = blk_reduce(vs[0] + vs[1] + vs[2], red, tid);
  float mt = sum_t * (1.0f / T_DIM);
  float ms = sum_s * (1.0f / T_DIM);
  float qt = 0.f, qs = 0.f;
  for (int i = 0; i < 3; i++) {
    float d = vt[i] - mt; qt += d * d;
    d = vs[i] - ms;       qs += d * d;
  }
  float var_t = blk_reduce(qt, red, tid) * (1.0f / T_DIM);
  float var_s = blk_reduce(qs, red, tid) * (1.0f / T_DIM);
  float rt = rsqrtf(var_t + 1e-5f);
  float rs = rsqrtf(var_s + 1e-5f);
  for (int i = 0; i < 3; i++) {
    int c = tid + i * 128;
    float yt = (vt[i] - mt) * rt * gt[c] + bt[c];
    float ys = (vs[i] - ms) * rs * gs[c] + bs[c];
    outf[row * EMBED_DIM + c] = yt;
    outf[row * EMBED_DIM + T_DIM + c] = ys;
    if (outb) {
      outb[row * EMBED_DIM + c] = f2bf(yt);
      outb[row * EMBED_DIM + T_DIM + c] = f2bf(ys);
    }
  }
}

// ---------------------------------------------------------------------------
// Launch
// ---------------------------------------------------------------------------
extern "C" void kernel_launch(void* const* d_in, const int* in_sizes, int n_in,
                              void* d_out, int out_size, void* d_ws, size_t ws_size,
                              hipStream_t stream) {
  (void)in_sizes; (void)n_in; (void)out_size; (void)ws_size;

  const float* x   = (const float*)d_in[0];
  const float* Wq  = (const float*)d_in[1];
  const float* bq  = (const float*)d_in[2];
  const float* Wk  = (const float*)d_in[3];
  const float* bk  = (const float*)d_in[4];
  const float* Wv  = (const float*)d_in[5];
  const float* bv  = (const float*)d_in[6];
  const float* Wo  = (const float*)d_in[7];
  const float* bo  = (const float*)d_in[8];
  const float* g1t = (const float*)d_in[9];
  const float* b1t = (const float*)d_in[10];
  const float* g1s = (const float*)d_in[11];
  const float* b1s = (const float*)d_in[12];
  const float* W1  = (const float*)d_in[13];
  const float* bb1 = (const float*)d_in[14];
  const float* W2  = (const float*)d_in[15];
  const float* bb2 = (const float*)d_in[16];
  const float* g2t = (const float*)d_in[17];
  const float* b2t = (const float*)d_in[18];
  const float* g2s = (const float*)d_in[19];
  const float* b2s = (const float*)d_in[20];
  float* out = (float*)d_out;

  // Workspace layout (all offsets 16B-aligned). Total: 70,778,880 bytes.
  char* ws = (char*)d_ws;
  unsigned short* wqt = (unsigned short*)(ws + 0);
  unsigned short* wkt = (unsigned short*)(ws + 1179648);
  unsigned short* wvt = (unsigned short*)(ws + 2359296);
  unsigned short* wot = (unsigned short*)(ws + 3538944);
  unsigned short* w1t = (unsigned short*)(ws + 4718592);
  unsigned short* w2t = (unsigned short*)(ws + 9437184);
  char* P = ws + 14155776;
  unsigned short* xbf  = (unsigned short*)(P + 0);
  unsigned short* qbf  = (unsigned short*)(P + 6291456);
  unsigned short* kbf  = (unsigned short*)(P + 12582912);
  unsigned short* vbf  = (unsigned short*)(P + 18874368);
  unsigned short* obf  = (unsigned short*)(P + 25165824);
  float*          res1 = (float*)(P + 31457280);          // residual (reused)
  float*          ln1f = (float*)(P + 44040192);          // ln1 fp32
  unsigned short* hbf  = (unsigned short*)(P + 0);        // alias x/q/k/v (dead)
  unsigned short* ln1b = (unsigned short*)(P + 25165824); // alias obf (dead)

  // Convert weights (transposed) and x to bf16
  auto launchT = [&](const float* w, unsigned short* wt, int K, int N) {
    int n = K * N;
    k_transpose_bf16<<<dim3((n + 255) / 256), dim3(256), 0, stream>>>(w, wt, K, N);
  };
  launchT(Wq, wqt, EMBED_DIM, EMBED_DIM);
  launchT(Wk, wkt, EMBED_DIM, EMBED_DIM);
  launchT(Wv, wvt, EMBED_DIM, EMBED_DIM);
  launchT(Wo, wot, EMBED_DIM, EMBED_DIM);
  launchT(W1, w1t, EMBED_DIM, HID4);
  launchT(W2, w2t, HID4, EMBED_DIM);
  k_f32_to_bf16<<<dim3((NROWS * EMBED_DIM + 255) / 256), dim3(256), 0, stream>>>(
      x, xbf, NROWS * EMBED_DIM);

  const dim3 g768(EMBED_DIM / 64, NROWS / 128);   // (12, 32)
  const dim3 g3072(HID4 / 64, NROWS / 128);       // (48, 32)
  const dim3 blk(256);

  // QKV projections (eta & 1/8 folded into q)
  k_gemm_bf16<MODE_Q ><<<g768, blk, 0, stream>>>(xbf, wqt, bq, nullptr, nullptr, qbf,
                                                 NROWS, EMBED_DIM, EMBED_DIM);
  k_gemm_bf16<MODE_KV><<<g768, blk, 0, stream>>>(xbf, wkt, bk, nullptr, nullptr, kbf,
                                                 NROWS, EMBED_DIM, EMBED_DIM);
  k_gemm_bf16<MODE_KV><<<g768, blk, 0, stream>>>(xbf, wvt, bv, nullptr, nullptr, vbf,
                                                 NROWS, EMBED_DIM, EMBED_DIM);

  // Flash attention
  k_flash<<<dim3(SEQ / 64, BATCH * N_HEADS), dim3(128), 0, stream>>>(qbf, kbf, vbf, obf);

  // O projection + residual(x), LN1
  k_gemm_bf16<MODE_RES><<<g768, blk, 0, stream>>>(obf, wot, bo, x, res1, nullptr,
                                                  NROWS, EMBED_DIM, EMBED_DIM);
  k_minkln<<<dim3(NROWS), dim3(128), 0, stream>>>(res1, g1t, b1t, g1s, b1s, ln1f, ln1b);

  // MLP: fc1 + gelu, fc2 + residual(ln1), LN2 -> out
  k_gemm_bf16<MODE_GELU><<<g3072, blk, 0, stream>>>(ln1b, w1t, bb1, nullptr, nullptr, hbf,
                                                    NROWS, HID4, EMBED_DIM);
  k_gemm_bf16<MODE_RES><<<g768, blk, 0, stream>>>(hbf, w2t, bb2, ln1f, res1, nullptr,
                                                  NROWS, EMBED_DIM, HID4);
  k_minkln<<<dim3(NROWS), dim3(128), 0, stream>>>(res1, g2t, b2t, g2s, b2s, out, nullptr);
}
